// EdgeConvEncoder_31748398252834
// MI455X (gfx1250) — matrix operand order repaired and checked by reading, not compiled, not run on hardware
//
#include <hip/hip_runtime.h>
#include <hip/hip_bf16.h>

typedef __attribute__((ext_vector_type(16))) _Float16 v16h;
typedef __attribute__((ext_vector_type(8)))  float    v8f;
typedef __attribute__((ext_vector_type(4)))  float    v4f;

#define WAVES_PER_BLOCK 4

// ---------------------------------------------------------------------------
// Zero-fill helper (grid-stride)
// ---------------------------------------------------------------------------
__global__ void zero_kernel(float* __restrict__ p, long long n) {
    long long i = (long long)blockIdx.x * blockDim.x + threadIdx.x;
    long long stride = (long long)gridDim.x * blockDim.x;
    for (; i < n; i += stride) p[i] = 0.0f;
}

// ---------------------------------------------------------------------------
// Repack fp32 row-major weights [K x 128] into WMMA B-fragment layout (f16).
// Fragment (kc, nt) covers K = kc*32..+31, N = nt*16..+15.
// Per ISA 7.12.2 (16-bit B 32x16): lane L -> N = L%16, halfK = L/16;
//   halfs[2v+j] = B[kc*32 + 16*(v/4) + 8*halfK + 2*(v%4) + j][N].
// Per conv layout (halfs): layer0 frags (kc<8,nt<8) at (kc*8+nt)*512,
// layers 1..3 at 32768 + (l-1)*16384 + (kc*8+nt)*512.  Conv stride 81920.
// ---------------------------------------------------------------------------
__global__ void repack_kernel(const float* __restrict__ w10, const float* __restrict__ w11,
                              const float* __restrict__ w12, const float* __restrict__ w13,
                              const float* __restrict__ w20, const float* __restrict__ w21,
                              const float* __restrict__ w22, const float* __restrict__ w23,
                              _Float16* __restrict__ wpack) {
    int tid = blockIdx.x * blockDim.x + threadIdx.x;
    if (tid >= 320 * 32) return;            // 2 convs * 160 frags * 32 lanes
    int lane = tid & 31;
    int f    = tid >> 5;
    int conv = f / 160;
    int fl   = f % 160;
    int l, kc, nt;
    long fragoff;
    if (fl < 64) {                           // layer 0: 8 kc x 8 nt
        l = 0; kc = fl >> 3; nt = fl & 7;
        fragoff = (long)(kc * 8 + nt) * 512;
    } else {                                 // layers 1..3: 4 kc x 8 nt each
        int t = fl - 64;
        l = 1 + t / 32;
        int rr = t % 32; kc = rr >> 3; nt = rr & 7;
        fragoff = 32768 + (long)(l - 1) * 16384 + (long)(kc * 8 + nt) * 512;
    }
    const float* W;
    if (conv == 0) W = (l == 0) ? w10 : (l == 1) ? w11 : (l == 2) ? w12 : w13;
    else           W = (l == 0) ? w20 : (l == 1) ? w21 : (l == 2) ? w22 : w23;

    int halfK = lane >> 4;
    int col   = nt * 16 + (lane & 15);
    _Float16* dstp = wpack + (long)conv * 81920 + fragoff + (long)lane * 16;
#pragma unroll
    for (int v = 0; v < 8; ++v) {
#pragma unroll
        for (int j = 0; j < 2; ++j) {
            int K = kc * 32 + 16 * (v >> 2) + 8 * halfK + 2 * (v & 3) + j;
            dstp[2 * v + j] = (_Float16)W[(long)K * 128 + col];
        }
    }
}

// ---------------------------------------------------------------------------
// A-fragment builder: 16-bit A 16x32 layout (ISA 7.12.2):
// lane L -> M = L%16, half = L/16; halfs[2v+j] = A[M][16*(v/4) + 8*half + 2*(v%4) + j]
// ---------------------------------------------------------------------------
__device__ __forceinline__ v16h load_a_frag(const _Float16 (*F)[256],
                                            int mlane, int half_id, int kcbase) {
    unsigned tmp[8];
#pragma unroll
    for (int v = 0; v < 8; ++v) {
        int K = kcbase + 16 * (v >> 2) + 8 * half_id + 2 * (v & 3);
        tmp[v] = *(const unsigned*)&F[mlane][K];   // two packed halfs, 4B aligned
    }
    return __builtin_bit_cast(v16h, tmp);
}

// ---------------------------------------------------------------------------
// Fused EdgeConv: gather concat(xi, xj-xi) -> 4-layer MLP (WMMA f16, f32 acc)
// -> atomic scatter-add to out[dst].  One wave per 16-edge tile.
// ---------------------------------------------------------------------------
__global__ __launch_bounds__(128) void edgeconv_kernel(
    const float* __restrict__ xin,        // [N,128] node features
    const int*   __restrict__ ei,         // [2,E] (src row 0, dst row 1)
    const _Float16* __restrict__ wpack,   // packed B fragments for this conv
    const float* __restrict__ bias0, const float* __restrict__ bias1,
    const float* __restrict__ bias2, const float* __restrict__ bias3,
    float* __restrict__ out,              // [N,128] accumulator
    long long E, int relu_in) {

    __shared__ _Float16 feat[WAVES_PER_BLOCK][16][256];
    __shared__ int      dstid[WAVES_PER_BLOCK][16];

    const int lane    = threadIdx.x & 31;
    const int wave    = threadIdx.x >> 5;
    const int mlane   = lane & 15;
    const int half_id = lane >> 4;
    const long long tile  = (long long)blockIdx.x * WAVES_PER_BLOCK + wave;
    const long long ebase = tile * 16;

    _Float16 (*F)[256] = feat[wave];
    int* dq = dstid[wave];

    // ---------------- gather: 16 edges, each row built by whole wave --------
    for (int r = 0; r < 16; ++r) {
        long long e = ebase + r;
        int sidx = 0, didx = 0;
        bool live = (e < E);
        if (live) { sidx = ei[e]; didx = ei[E + e]; }
        if (lane == 0) dq[r] = didx;

        v4f xi = {}, xj = {};
        if (live) {
            xi = *(const v4f*)(xin + (long long)didx * 128 + lane * 4);
            xj = *(const v4f*)(xin + (long long)sidx * 128 + lane * 4);
        }
        if (relu_in) {
#pragma unroll
            for (int k = 0; k < 4; ++k) {
                xi[k] = fmaxf(xi[k], 0.0f);
                xj[k] = fmaxf(xj[k], 0.0f);
            }
        }
#pragma unroll
        for (int k = 0; k < 4; ++k) {
            F[r][lane * 4 + k]       = (_Float16)xi[k];
            F[r][128 + lane * 4 + k] = (_Float16)(xj[k] - xi[k]);
        }
    }
    __builtin_amdgcn_wave_barrier();   // LDS is in-order per wave; pin schedule

    // ---------------- layer 0: [16x256] x [256x128] -------------------------
    {
        v8f acc[8] = {};
        for (int kc = 0; kc < 8; ++kc) {
            v16h a = load_a_frag(F, mlane, half_id, kc * 32);
            const _Float16* wb = wpack + (long)(kc * 8) * 512 + (long)lane * 16;
#pragma unroll
            for (int nt = 0; nt < 8; ++nt) {
                v16h b = *(const v16h*)(wb + (long)nt * 512);
                acc[nt] = __builtin_amdgcn_wmma_f32_16x16x32_f16(
                    false, a, false, b, (short)0, acc[nt], false, false);
            }
        }
        __builtin_amdgcn_wave_barrier();
#pragma unroll
        for (int nt = 0; nt < 8; ++nt) {
            float bb = bias0[nt * 16 + mlane];
#pragma unroll
            for (int r = 0; r < 8; ++r) {
                float v = fmaxf(acc[nt][r] + bb, 0.0f);          // ReLU
                F[r + 8 * half_id][nt * 16 + mlane] = (_Float16)v;
            }
        }
        __builtin_amdgcn_wave_barrier();
    }

    // ---------------- layers 1..3: [16x128] x [128x128] ---------------------
    for (int l = 1; l <= 3; ++l) {
        const _Float16* wl = wpack + 32768 + (long)(l - 1) * 16384;
        const float* bl = (l == 1) ? bias1 : (l == 2) ? bias2 : bias3;

        v8f acc[8] = {};
        for (int kc = 0; kc < 4; ++kc) {
            v16h a = load_a_frag(F, mlane, half_id, kc * 32);
            const _Float16* wb = wl + (long)(kc * 8) * 512 + (long)lane * 16;
#pragma unroll
            for (int nt = 0; nt < 8; ++nt) {
                v16h b = *(const v16h*)(wb + (long)nt * 512);
                acc[nt] = __builtin_amdgcn_wmma_f32_16x16x32_f16(
                    false, a, false, b, (short)0, acc[nt], false, false);
            }
        }
        __builtin_amdgcn_wave_barrier();

        if (l < 3) {                          // bias + ReLU, back to LDS
#pragma unroll
            for (int nt = 0; nt < 8; ++nt) {
                float bb = bl[nt * 16 + mlane];
#pragma unroll
                for (int r = 0; r < 8; ++r) {
                    float v = fmaxf(acc[nt][r] + bb, 0.0f);
                    F[r + 8 * half_id][nt * 16 + mlane] = (_Float16)v;
                }
            }
            __builtin_amdgcn_wave_barrier();
        } else {                              // final layer: scatter-add
#pragma unroll
            for (int nt = 0; nt < 8; ++nt) {
                float bb = bl[nt * 16 + mlane];
#pragma unroll
                for (int r = 0; r < 8; ++r) {
                    int row = r + 8 * half_id;
                    long long e = ebase + row;
                    if (e < E) {
                        int node = dq[row];
                        float* dst = out + (long long)node * 128 + nt * 16 + mlane;
                        __hip_atomic_fetch_add(dst, acc[nt][r] + bb,
                                               __ATOMIC_RELAXED,
                                               __HIP_MEMORY_SCOPE_AGENT);
                    }
                }
            }
        }
    }
}

// ---------------------------------------------------------------------------
// Host entry
// ---------------------------------------------------------------------------
extern "C" void kernel_launch(void* const* d_in, const int* in_sizes, int n_in,
                              void* d_out, int out_size, void* d_ws, size_t ws_size,
                              hipStream_t stream) {
    (void)n_in; (void)out_size; (void)ws_size;

    const float* x  = (const float*)d_in[0];
    const int*   ei = (const int*)d_in[1];
    // params1: (W,b) x4 at indices 2..9 ; params2 at 10..17
    const float* w10 = (const float*)d_in[2];  const float* b10 = (const float*)d_in[3];
    const float* w11 = (const float*)d_in[4];  const float* b11 = (const float*)d_in[5];
    const float* w12 = (const float*)d_in[6];  const float* b12 = (const float*)d_in[7];
    const float* w13 = (const float*)d_in[8];  const float* b13 = (const float*)d_in[9];
    const float* w20 = (const float*)d_in[10]; const float* b20 = (const float*)d_in[11];
    const float* w21 = (const float*)d_in[12]; const float* b21 = (const float*)d_in[13];
    const float* w22 = (const float*)d_in[14]; const float* b22 = (const float*)d_in[15];
    const float* w23 = (const float*)d_in[16]; const float* b23 = (const float*)d_in[17];
    float* out = (float*)d_out;

    const long long Nn = in_sizes[0] / 128;
    const long long E  = in_sizes[1] / 2;

    // workspace: h [N,128] f32, then packed weights (2 convs x 81920 halfs)
    float* h = (float*)d_ws;
    size_t hbytes = (size_t)Nn * 128 * sizeof(float);
    size_t woff   = (hbytes + 255) & ~(size_t)255;
    _Float16* wpack1 = (_Float16*)((char*)d_ws + woff);
    _Float16* wpack2 = (_Float16*)((char*)d_ws + woff + (size_t)81920 * sizeof(_Float16));

    const long long nfeat = Nn * 128;
    zero_kernel<<<2048, 256, 0, stream>>>(h, nfeat);
    zero_kernel<<<2048, 256, 0, stream>>>(out, nfeat);

    repack_kernel<<<(320 * 32 + 255) / 256, 256, 0, stream>>>(
        w10, w11, w12, w13, w20, w21, w22, w23, wpack1);

    const long long tiles  = (E + 15) / 16;
    const int       blocks = (int)((tiles + WAVES_PER_BLOCK - 1) / WAVES_PER_BLOCK);

    // conv1: x -> h (segment-sum of MLP over edges)
    edgeconv_kernel<<<blocks, 128, 0, stream>>>(
        x, ei, wpack1, b10, b11, b12, b13, h, E, /*relu_in=*/0);

    // conv2: relu(h) -> out
    edgeconv_kernel<<<blocks, 128, 0, stream>>>(
        h, ei, wpack2, b20, b21, b22, b23, out, E, /*relu_in=*/1);
}
